// EmbGCNEncoder_24051816858276
// MI455X (gfx1250) — matrix operand order repaired and checked by reading, not compiled, not run on hardware
//
#include <hip/hip_runtime.h>
#include <hip/hip_bf16.h>

typedef __attribute__((ext_vector_type(16))) __bf16 v16bf;
typedef __attribute__((ext_vector_type(8)))  float  v8f;

// ---------------------------------------------------------------- utilities

__global__ void zero4_kernel(float4* __restrict__ p, long n4) {
    long i = (long)blockIdx.x * blockDim.x + threadIdx.x;
    if (i < n4) p[i] = make_float4(0.f, 0.f, 0.f, 0.f);
}

__global__ void degree_kernel(const int* __restrict__ src, const int* __restrict__ dst,
                              float* __restrict__ deg_out, float* __restrict__ deg_in, int E) {
    int e = blockIdx.x * blockDim.x + threadIdx.x;
    if (e < E) {
        atomicAdd(&deg_out[src[e]], 1.0f);
        atomicAdd(&deg_in [dst[e]], 1.0f);
    }
}

// in place: d -> rsqrt(max(d,1))
__global__ void norm_kernel(float* __restrict__ p, int n) {
    int i = blockIdx.x * blockDim.x + threadIdx.x;
    if (i < n) p[i] = rsqrtf(fmaxf(p[i], 1.0f));
}

// x[i,:] = emb[batch[i],:]   (one float4 per thread, 32 threads per row)
__global__ void gather_kernel(const int* __restrict__ batch, const float* __restrict__ emb,
                              float* __restrict__ x, int N) {
    int i = blockIdx.x * blockDim.x + threadIdx.x;
    if (i < N * 32) {
        int node = i >> 5, q = i & 31;
        reinterpret_cast<float4*>(x)[(size_t)node * 32 + q] =
            reinterpret_cast<const float4*>(emb)[(size_t)batch[node] * 32 + q];
    }
}

// agg[dst[e],:] += x[src[e],:] * norm_src[src[e]]   (one wave per edge)
__global__ void spmm_kernel(const int* __restrict__ src, const int* __restrict__ dst,
                            const float* __restrict__ x, const float* __restrict__ norm_src,
                            float* __restrict__ agg, int E) {
    int e = blockIdx.x * (blockDim.x >> 5) + (threadIdx.x >> 5);
    if (e >= E) return;
    int lane = threadIdx.x & 31;
    int s = src[e], d = dst[e];
    float ns = norm_src[s];
    float4 v = reinterpret_cast<const float4*>(x + (size_t)s * 128)[lane];
    float* o = agg + (size_t)d * 128 + lane * 4;
    atomicAdd(o + 0, v.x * ns);
    atomicAdd(o + 1, v.y * ns);
    atomicAdd(o + 2, v.z * ns);
    atomicAdd(o + 3, v.w * ns);
}

// ---------------------------------------------------------------- WMMA GEMM
// out[m,:] = relu( (Ain[m,:] * norm_dst[m]) @ W + bias ),  W is [128,128] row-major.
// 128 threads (4 waves); each wave computes a 16x128 output strip.
__global__ void __launch_bounds__(128)
gemm_bias_relu_kernel(const float* __restrict__ Ain, const float* __restrict__ norm_dst,
                      const float* __restrict__ W, const float* __restrict__ bias,
                      float* __restrict__ out, int N) {
    __shared__ __bf16 Wt[128][136];      // W transposed: Wt[n][k] = W[k][n]
    __shared__ __bf16 As[4][16][136];    // per-wave A tile (scaled by norm_dst)

    const int tid  = threadIdx.x;
    const int wave = tid >> 5;
    const int lane = tid & 31;
    const int hl   = lane >> 4;          // lane half (0/1)
    const int lr   = lane & 15;          // row (A) / col (B,C) within tile

    // stage W^T in bf16 (coalesced global reads)
    for (int i = tid; i < 128 * 128; i += 128) {
        int k = i >> 7, n = i & 127;
        Wt[n][k] = (__bf16)W[i];
    }

    // stage this wave's 16x128 A tile, scaled by norm_dst, zero-padded past N
    const int rowBase = blockIdx.x * 64 + wave * 16;
    for (int r = 0; r < 16; ++r) {
        int gr = rowBase + r;
        float4 v = make_float4(0.f, 0.f, 0.f, 0.f);
        float nd = 0.f;
        if (gr < N) {
            nd = norm_dst[gr];
            v = reinterpret_cast<const float4*>(Ain + (size_t)gr * 128)[lane];
        }
        __bf16* dp = &As[wave][r][lane * 4];
        dp[0] = (__bf16)(v.x * nd);
        dp[1] = (__bf16)(v.y * nd);
        dp[2] = (__bf16)(v.z * nd);
        dp[3] = (__bf16)(v.w * nd);
    }
    __syncthreads();

    v8f acc[8] = {};

    for (int kc = 0; kc < 4; ++kc) {
        const int k0 = kc * 32;
        // A fragment per ISA 16-bit 16x32 layout:
        //  VGPR j (j<4): K = 8*hl + 2j, +1 ; VGPR 4+j: K = 16 + 8*hl + 2j, +1
        union { v16bf v; unsigned u[8]; } A;
        const __bf16* arow = &As[wave][lr][0];
        #pragma unroll
        for (int j = 0; j < 4; ++j) {
            A.u[j]     = *reinterpret_cast<const unsigned*>(arow + k0 + 8 * hl + 2 * j);
            A.u[4 + j] = *reinterpret_cast<const unsigned*>(arow + k0 + 16 + 8 * hl + 2 * j);
        }
        #pragma unroll
        for (int t = 0; t < 8; ++t) {
            // B fragment: col = lr, VGPR j (j<4) holds K = 8*hl + 2j, +1; VGPR 4+j: K = 16 + 8*hl + 2j, +1 (within this k-chunk)
            const __bf16* brow = &Wt[t * 16 + lr][0];
            union { v16bf v; unsigned u[8]; } B;
            #pragma unroll
            for (int j = 0; j < 8; ++j)
                B.u[j] = *reinterpret_cast<const unsigned*>(brow + k0 + 16 * (j >> 2) + 8 * hl + 2 * (j & 3));
            acc[t] = __builtin_amdgcn_wmma_f32_16x16x32_bf16(
                         false, A.v, false, B.v, (short)0, acc[t], false, false);
        }
    }

    // epilogue: C/D layout — VGPR r: row = r + 8*hl, col = lr (per 16x16 tile)
    #pragma unroll
    for (int t = 0; t < 8; ++t) {
        const int col = t * 16 + lr;
        const float bv = bias[col];
        #pragma unroll
        for (int r = 0; r < 8; ++r) {
            int grow = rowBase + 8 * hl + r;
            if (grow < N) {
                float v = acc[t][r] + bv;
                out[(size_t)grow * 128 + col] = v > 0.f ? v : 0.f;
            }
        }
    }
}

// ---------------------------------------------------------------- launcher

extern "C" void kernel_launch(void* const* d_in, const int* in_sizes, int n_in,
                              void* d_out, int out_size, void* d_ws, size_t ws_size,
                              hipStream_t stream) {
    const int*   batch = (const int*)  d_in[0];
    const int*   src   = (const int*)  d_in[1];
    const int*   dst   = (const int*)  d_in[2];
    const float* emb   = (const float*)d_in[3];
    const float* W1    = (const float*)d_in[4];
    const float* b1    = (const float*)d_in[5];
    const float* W2    = (const float*)d_in[6];
    const float* b2    = (const float*)d_in[7];
    float* out = (float*)d_out;

    const int N = in_sizes[0];   // 50000 nodes
    const int E = in_sizes[1];   // 500000 edges

    float* ws      = (float*)d_ws;
    float* deg_out = ws;                          // [N] -> norm_src after norm_kernel
    float* deg_in  = ws + N;                      // [N] -> norm_dst after norm_kernel
    float* xa      = ws + 2 * (size_t)N;          // [N,128]
    float* agg     = xa + (size_t)N * 128;        // [N,128]

    const int TB = 256;
    const long feat4 = (long)N * 32;              // N*128/4 float4s
    const int gemmGrid = (N + 63) / 64;
    const int spmmGrid = (E + (TB / 32) - 1) / (TB / 32);

    // degrees + norms
    zero4_kernel<<<((2 * N / 4) + TB - 1) / TB, TB, 0, stream>>>((float4*)deg_out, 2L * N / 4);
    degree_kernel<<<(E + TB - 1) / TB, TB, 0, stream>>>(src, dst, deg_out, deg_in, E);
    norm_kernel<<<(2 * N + TB - 1) / TB, TB, 0, stream>>>(deg_out, 2 * N);

    // x = emb[batch]
    gather_kernel<<<((long)N * 32 + TB - 1) / TB, TB, 0, stream>>>(batch, emb, xa, N);

    // layer 1: agg = scatter(x*norm_src); xa = relu(agg*norm_dst @ W1 + b1)
    zero4_kernel<<<(feat4 + TB - 1) / TB, TB, 0, stream>>>((float4*)agg, feat4);
    spmm_kernel<<<spmmGrid, TB, 0, stream>>>(src, dst, xa, deg_out, agg, E);
    gemm_bias_relu_kernel<<<gemmGrid, 128, 0, stream>>>(agg, deg_in, W1, b1, xa, N);

    // layer 2: same, writing d_out
    zero4_kernel<<<(feat4 + TB - 1) / TB, TB, 0, stream>>>((float4*)agg, feat4);
    spmm_kernel<<<spmmGrid, TB, 0, stream>>>(src, dst, xa, deg_out, agg, E);
    gemm_bias_relu_kernel<<<gemmGrid, 128, 0, stream>>>(agg, deg_in, W2, b2, out, N);
}
